// ViT_backbone_36447092474458
// MI455X (gfx1250) — matrix operand order repaired and checked by reading, not compiled, not run on hardware
//
#include <hip/hip_runtime.h>
#include <hip/hip_bf16.h>

#define D_EMB   1024
#define NTOK    257
#define BATCH   64
#define NLAYER  8
#define NHEAD   8
#define HDIM    128
#define DFF     4096
#define MTOK    (BATCH * NTOK)   // 16448 = 1028*16 = 257*64

typedef __attribute__((ext_vector_type(16))) __bf16 bf16x16;
typedef __attribute__((ext_vector_type(8)))  float  floatx8;

union FragAB { bf16x16 v; uint4 q[2]; unsigned short u[16]; };
union FragC  { floatx8 v; float f[8]; };

__device__ inline unsigned short bf16_bits(float f) {
  unsigned u = __float_as_uint(f);
  return (unsigned short)((u + 0x7FFFu + ((u >> 16) & 1u)) >> 16);
}

__device__ inline floatx8 wmma_bf16(bf16x16 a, bf16x16 b, floatx8 c) {
  // D = A(16x32 bf16) * B(32x16 bf16) + C(16x16 f32)
  return __builtin_amdgcn_wmma_f32_16x16x32_bf16(false, a, false, b, (short)0, c, false, false);
}

// gfx1250 async DMA: global -> LDS, 16 bytes per lane, tracked by ASYNCcnt.
__device__ inline void async_g2l_b128(unsigned lds_byte_off, const void* gaddr) {
  asm volatile("global_load_async_to_lds_b128 %0, %1, off"
               :: "v"(lds_byte_off), "v"(gaddr) : "memory");
}
__device__ inline void wait_async0() {
  asm volatile("s_wait_asynccnt 0x0" ::: "memory");
}
// low 32 bits of a flat shared pointer == LDS byte offset (aperture mapping)
__device__ inline unsigned lds_off32(const void* p) {
  return (unsigned)(size_t)p;
}

// ---------------------------------------------------------------------------
// Weight re-encode: f32 [K,N] row-major -> bf16 [N,K] (transposed so WMMA
// B-fragments are K-contiguous and can be async-DMA'd straight into LDS).
// Grid covers N*K elements; writes are coalesced along K.
// ---------------------------------------------------------------------------
__global__ __launch_bounds__(256)
void cvt_t_bf16_kernel(const float* __restrict__ in, unsigned short* __restrict__ out,
                       int K, int N) {
  long idx = (long)blockIdx.x * 256 + threadIdx.x;   // over N*K
  long n = idx / K, k = idx - n * K;
  out[idx] = bf16_bits(in[k * (long)N + n]);
}

// ---------------------------------------------------------------------------
// Patch embedding + cls token + positional embedding -> t (fp32 [MTOK, 1024])
// ---------------------------------------------------------------------------
__global__ __launch_bounds__(256)
void patch_embed_kernel(const float* __restrict__ x, const float* __restrict__ cw,
                        const float* __restrict__ cb, const float* __restrict__ cls,
                        const float* __restrict__ pos, float* __restrict__ t) {
  int tok = blockIdx.x % NTOK;
  int bb  = blockIdx.x / NTOK;
  int tid = threadIdx.x;
  long obase = ((long)bb * NTOK + tok) * D_EMB;
  if (tok == 0) {
    #pragma unroll
    for (int i = 0; i < 4; i++) {
      int d = tid + i * 256;
      t[obase + d] = cls[d] + pos[d];
    }
    return;
  }
  int p  = tok - 1;
  int py = p >> 4, px = p & 15;
  const float* xb = x + (((long)bb * 64 + py * 4) * 64 + px * 4) * 3;
  #pragma unroll
  for (int i = 0; i < 4; i++) {
    int d = tid + i * 256;
    float s = cb[d];
    for (int iy = 0; iy < 4; iy++)
      for (int ix = 0; ix < 4; ix++)
        for (int c = 0; c < 3; c++)
          s += xb[(iy * 64 + ix) * 3 + c] * cw[((iy * 4 + ix) * 3 + c) * D_EMB + d];
    t[obase + d] = s + pos[(long)tok * D_EMB + d];
  }
}

// ---------------------------------------------------------------------------
// LayerNorm over D=1024, output bf16. One block (256 threads) per row.
// ---------------------------------------------------------------------------
__global__ __launch_bounds__(256)
void layernorm_bf16_kernel(const float* __restrict__ x, const float* __restrict__ g,
                           const float* __restrict__ b, unsigned short* __restrict__ out) {
  __shared__ float red[256];
  long row = blockIdx.x;
  const float* xr = x + row * D_EMB;
  int tid = threadIdx.x;
  float v0[4];
  float s = 0.f;
  #pragma unroll
  for (int i = 0; i < 4; i++) { v0[i] = xr[tid + i * 256]; s += v0[i]; }
  red[tid] = s; __syncthreads();
  for (int o = 128; o > 0; o >>= 1) { if (tid < o) red[tid] += red[tid + o]; __syncthreads(); }
  float mean = red[0] * (1.0f / 1024.0f);
  __syncthreads();
  float s2 = 0.f;
  #pragma unroll
  for (int i = 0; i < 4; i++) { float d = v0[i] - mean; s2 += d * d; }
  red[tid] = s2; __syncthreads();
  for (int o = 128; o > 0; o >>= 1) { if (tid < o) red[tid] += red[tid + o]; __syncthreads(); }
  float rstd = rsqrtf(red[0] * (1.0f / 1024.0f) + 1e-6f);
  #pragma unroll
  for (int i = 0; i < 4; i++) {
    int c = tid + i * 256;
    out[row * D_EMB + c] = bf16_bits((v0[i] - mean) * rstd * g[c] + b[c]);
  }
}

// ---------------------------------------------------------------------------
// bf16 WMMA GEMM: C[M,Nc] = scale*(A[M,K] @ W[K,Nc]) + bias (+GELU)(+res).
// Weights pre-transposed: Wt[N,K]. M tile 64, N tile 128, K step 32.
// Double-buffered LDS tiles filled by GLOBAL_LOAD_ASYNC_TO_LDS_B128 so the
// DMA for step i+1 overlaps WMMA on step i. 8 waves / block.
// ---------------------------------------------------------------------------
#define GMT 64
#define GNT 128
#define GKT 32

__global__ __launch_bounds__(256)
void gemm_bf16_kernel(const unsigned short* __restrict__ A,
                      const unsigned short* __restrict__ Wt,   // [Nc, K] bf16
                      const float* __restrict__ bias,
                      const float* __restrict__ res,
                      float* __restrict__ out32,
                      unsigned short* __restrict__ out16,
                      int Nc, int K, float scale, int act) {
  __shared__ __align__(16) unsigned short lds_a[2][GMT * GKT];   // [m][k]
  __shared__ __align__(16) unsigned short lds_bt[2][GNT * GKT];  // [n][k]

  int tid  = threadIdx.x;
  int lane = tid & 31;
  int wave = tid >> 5;
  int wm   = wave >> 1;   // 0..3 : 16-row stripe
  int wn   = wave & 1;    // 0..1 : 64-col stripe
  long m0  = (long)blockIdx.x * GMT;
  int  n0  = blockIdx.y * GNT;

  // cooperative async-load coordinates (loop invariant)
  int ar = tid >> 2;            // 0..63   (A row)
  int ac = (tid & 3) * 8;       // 0..24   (A k-chunk)
  int bn = tid >> 1;            // 0..127  (B row = output column)
  int bc = (tid & 1) * 16;      // 0 or 16 (B k-chunk base, two b128 each)

  const unsigned short* gA = A  + (m0 + ar) * (long)K + ac;
  const unsigned short* gB = Wt + (long)(n0 + bn) * K + bc;
  unsigned laA = lds_off32(&lds_a[0][0])  + (unsigned)((ar * GKT + ac) * 2);
  unsigned laB = lds_off32(&lds_bt[0][0]) + (unsigned)((bn * GKT + bc) * 2);
  const unsigned strA = GMT * GKT * 2;   // bytes between A buffers
  const unsigned strB = GNT * GKT * 2;   // bytes between B buffers

  FragC acc[4];
  #pragma unroll
  for (int i = 0; i < 4; i++)
    #pragma unroll
    for (int j = 0; j < 8; j++) acc[i].f[j] = 0.0f;

  int kb8  = (lane >> 4) << 3;   // A-frag K base
  int kb16 = (lane >> 4) << 4;   // unused (B-frag also uses kb8)
  int mfr  = lane & 15;

  auto issue = [&](int k0, int buf) {
    async_g2l_b128(laA + (unsigned)buf * strA, gA + k0);
    async_g2l_b128(laB + (unsigned)buf * strB, gB + k0);
    async_g2l_b128(laB + (unsigned)buf * strB + 16, gB + k0 + 8);
  };

  issue(0, 0);
  wait_async0();
  __syncthreads();

  int cur = 0;
  for (int k0 = 0; k0 < K; k0 += GKT) {
    if (k0 + GKT < K) issue(k0 + GKT, cur ^ 1);   // overlap DMA with WMMA

    FragAB fa;
    {
      const unsigned short* p = &lds_a[cur][(wm * 16 + mfr) * GKT + kb8];
      fa.q[0] = *(const uint4*)p;
      fa.q[1] = *(const uint4*)(p + 16);
    }
    #pragma unroll
    for (int nt = 0; nt < 4; nt++) {
      FragAB fb;
      const unsigned short* p = &lds_bt[cur][(wn * 64 + nt * 16 + mfr) * GKT + kb8];
      fb.q[0] = *(const uint4*)p;
      fb.q[1] = *(const uint4*)(p + 16);
      acc[nt].v = wmma_bf16(fa.v, fb.v, acc[nt].v);
    }

    wait_async0();       // my next-tile DMA has landed in LDS
    __syncthreads();     // everyone's DMA done & everyone done reading cur
    cur ^= 1;
  }

  // epilogue: C layout -> lane gives column, VGPR index gives row
  int rbase = wm * 16 + ((lane >> 4) << 3);
  int cbase = n0 + wn * 64 + mfr;
  #pragma unroll
  for (int nt = 0; nt < 4; nt++) {
    int c = cbase + nt * 16;
    float bv = bias ? bias[c] : 0.0f;
    #pragma unroll
    for (int vv = 0; vv < 8; vv++) {
      long r = m0 + rbase + vv;
      long o = r * Nc + c;
      float x = acc[nt].f[vv] * scale + bv;
      if (act == 1) x = 0.5f * x * (1.0f + erff(x * 0.70710678118f));  // exact GELU
      if (res)   x += res[o];
      if (out32) out32[o] = x;
      if (out16) out16[o] = bf16_bits(x);
    }
  }
}

// ---------------------------------------------------------------------------
// Fused attention: one wave per (q-tile, head, batch).
// scores = Q Kt (WMMA, keys padded to 288 & masked), softmax (fp32, LDS),
// probs -> bf16, out = P V (WMMA, V staged K-transposed in LDS).
// ---------------------------------------------------------------------------
#define NPAD 288   // 9 * 32

__global__ __launch_bounds__(32)
void attn_fused_kernel(const unsigned short* __restrict__ q,
                       const unsigned short* __restrict__ k,
                       const unsigned short* __restrict__ v,
                       unsigned short* __restrict__ y) {
  __shared__ __align__(16) float          lds_s[16 * NPAD];
  __shared__ __align__(16) unsigned short lds_p[16 * NPAD];
  __shared__ __align__(16) unsigned short lds_vt[HDIM * 32];  // [dim][token]
  __shared__ float lds_mx[16], lds_sum[16];

  int lane = threadIdx.x;
  int qt = blockIdx.x;   // 0..16
  int h  = blockIdx.y;   // 0..7
  int bb = blockIdx.z;   // 0..63

  int m    = lane & 15;
  int kb8  = (lane >> 4) << 3;
  int kb16 = (lane >> 4) << 4;

  // Q fragments for 4 K-steps (d = 0..127), rows clamped (invalid rows unused)
  int qrow = qt * 16 + m; if (qrow > NTOK - 1) qrow = NTOK - 1;
  const unsigned short* qp = q + ((long)bb * NTOK + qrow) * D_EMB + h * HDIM;
  FragAB fq[4];
  #pragma unroll
  for (int ks = 0; ks < 4; ks++) {
    const unsigned short* p = qp + ks * 32 + kb8;
    fq[ks].q[0] = *(const uint4*)p;
    fq[ks].q[1] = *(const uint4*)(p + 16);
  }

  // scores: 17 key tiles of 16
  for (int nt = 0; nt < 17; nt++) {
    FragC acc;
    #pragma unroll
    for (int j = 0; j < 8; j++) acc.f[j] = 0.0f;
    int key  = nt * 16 + m;
    int keyc = key > NTOK - 1 ? NTOK - 1 : key;
    const unsigned short* kp = k + ((long)bb * NTOK + keyc) * D_EMB + h * HDIM;
    #pragma unroll
    for (int ks = 0; ks < 4; ks++) {
      FragAB fb;   // B = K^T : fixed column (key), contiguous d
      const unsigned short* p = kp + ks * 32 + kb8;
      fb.q[0] = *(const uint4*)p;
      fb.q[1] = *(const uint4*)(p + 16);
      acc.v = wmma_bf16(fq[ks].v, fb.v, acc.v);
    }
    int rb = (lane >> 4) << 3;
    #pragma unroll
    for (int vv = 0; vv < 8; vv++)
      lds_s[(rb + vv) * NPAD + key] = (key < NTOK) ? acc.f[vv] : -1e30f;
  }

  // row softmax stats (lanes 0..15, one row each)
  if (lane < 16) {
    float mx = -1e30f;
    for (int c = 0; c < NTOK; c++) { float sv = lds_s[lane * NPAD + c]; mx = sv > mx ? sv : mx; }
    float sum = 0.0f;
    for (int c = 0; c < NTOK; c++) sum += __expf(lds_s[lane * NPAD + c] - mx);
    lds_mx[lane] = mx; lds_sum[lane] = 1.0f / sum;
  }
  __syncthreads();
  // normalized probs -> bf16, pad columns zeroed
  for (int u = lane; u < 16 * NPAD; u += 32) {
    int r = u / NPAD, c = u % NPAD;
    float pr = (c < NTOK) ? __expf(lds_s[r * NPAD + c] - lds_mx[r]) * lds_sum[r] : 0.0f;
    lds_p[r * NPAD + c] = bf16_bits(pr);
  }
  __syncthreads();

  // out = P[16,288] @ V[288,128] : 9 K-steps, V tile staged transposed in LDS
  FragC av[8];
  #pragma unroll
  for (int i = 0; i < 8; i++)
    #pragma unroll
    for (int j = 0; j < 8; j++) av[i].f[j] = 0.0f;

  for (int ks = 0; ks < 9; ks++) {
    #pragma unroll
    for (int i = 0; i < 16; i++) {
      int u  = i * 32 + lane;       // 0..511
      int r  = u >> 4;              // token in tile 0..31
      int c8 = (u & 15) * 8;        // dim 0..120
      int tokg = ks * 32 + r;
      uint4 d4 = make_uint4(0u, 0u, 0u, 0u);
      if (tokg < NTOK)
        d4 = *(const uint4*)(v + ((long)bb * NTOK + tokg) * D_EMB + h * HDIM + c8);
      const unsigned short* pd = (const unsigned short*)&d4;
      #pragma unroll
      for (int e = 0; e < 8; e++) lds_vt[(c8 + e) * 32 + r] = pd[e];
    }
    __syncthreads();
    FragAB fa;
    {
      const unsigned short* p = lds_p + m * NPAD + ks * 32 + kb8;
      fa.q[0] = *(const uint4*)p;
      fa.q[1] = *(const uint4*)(p + 16);
    }
    #pragma unroll
    for (int nt = 0; nt < 8; nt++) {
      FragAB fb;
      const unsigned short* p = lds_vt + (nt * 16 + m) * 32 + kb8;
      fb.q[0] = *(const uint4*)p;
      fb.q[1] = *(const uint4*)(p + 16);
      av[nt].v = wmma_bf16(fa.v, fb.v, av[nt].v);
    }
    __syncthreads();
  }

  // store y (bf16), head-contiguous layout [M, 1024]
  int rb = (lane >> 4) << 3;
  #pragma unroll
  for (int nt = 0; nt < 8; nt++) {
    #pragma unroll
    for (int vv = 0; vv < 8; vv++) {
      int qr = qt * 16 + rb + vv;
      if (qr < NTOK)
        y[((long)bb * NTOK + qr) * D_EMB + h * HDIM + nt * 16 + m] = bf16_bits(av[nt].f[vv]);
    }
  }
}

// ---------------------------------------------------------------------------
// Extract cls rows: out[b, :] = t[b*257, :]
// ---------------------------------------------------------------------------
__global__ __launch_bounds__(256)
void extract_cls_kernel(const float* __restrict__ t, float* __restrict__ out) {
  int b = blockIdx.x;
  #pragma unroll
  for (int i = 0; i < 4; i++) {
    int c = threadIdx.x + i * 256;
    out[(long)b * D_EMB + c] = t[(long)b * NTOK * D_EMB + c];
  }
}

// ---------------------------------------------------------------------------
extern "C" void kernel_launch(void* const* d_in, const int* in_sizes, int n_in,
                              void* d_out, int out_size, void* d_ws, size_t ws_size,
                              hipStream_t stream) {
  (void)in_sizes; (void)n_in; (void)out_size; (void)ws_size;
  const float* x        = (const float*)d_in[0];
  const float* conv_w   = (const float*)d_in[1];
  const float* conv_b   = (const float*)d_in[2];
  const float* cls_tok  = (const float*)d_in[3];
  const float* pos_emb  = (const float*)d_in[4];
  const float* ln1_g    = (const float*)d_in[5];
  const float* ln1_b    = (const float*)d_in[6];
  const float* wq       = (const float*)d_in[7];
  const float* wk       = (const float*)d_in[8];
  const float* wv       = (const float*)d_in[9];
  const float* proj_w   = (const float*)d_in[10];
  const float* proj_b   = (const float*)d_in[11];
  const float* ln2_g    = (const float*)d_in[12];
  const float* ln2_b    = (const float*)d_in[13];
  const float* mlp1_w   = (const float*)d_in[14];
  const float* mlp1_b   = (const float*)d_in[15];
  const float* mlp2_w   = (const float*)d_in[16];
  const float* mlp2_b   = (const float*)d_in[17];
  float* out = (float*)d_out;

  // workspace carve-out
  size_t off = 0;
  auto carve = [&](size_t bytes) -> char* {
    char* p = (char*)d_ws + off;
    off += (bytes + 255) & ~(size_t)255;
    return p;
  };
  float*          t   = (float*)carve((size_t)MTOK * D_EMB * 4);
  unsigned short* xn  = (unsigned short*)carve((size_t)MTOK * D_EMB * 2);
  unsigned short* qb  = (unsigned short*)carve((size_t)MTOK * D_EMB * 2);
  unsigned short* kb  = (unsigned short*)carve((size_t)MTOK * D_EMB * 2);
  unsigned short* vb  = (unsigned short*)carve((size_t)MTOK * D_EMB * 2);
  unsigned short* yb  = (unsigned short*)carve((size_t)MTOK * D_EMB * 2);
  unsigned short* hb  = (unsigned short*)carve((size_t)MTOK * DFF * 2);
  unsigned short* wqb = (unsigned short*)carve((size_t)NLAYER * D_EMB * D_EMB * 2);
  unsigned short* wkb = (unsigned short*)carve((size_t)NLAYER * D_EMB * D_EMB * 2);
  unsigned short* wvb = (unsigned short*)carve((size_t)NLAYER * D_EMB * D_EMB * 2);
  unsigned short* wpb = (unsigned short*)carve((size_t)NLAYER * D_EMB * D_EMB * 2);
  unsigned short* m1b = (unsigned short*)carve((size_t)NLAYER * D_EMB * DFF * 2);
  unsigned short* m2b = (unsigned short*)carve((size_t)NLAYER * DFF * D_EMB * 2);

  // weight re-encode to bf16, transposed to [N,K] per layer
  auto cvt_t = [&](const float* src, unsigned short* dst, int K, int N) {
    for (int i = 0; i < NLAYER; i++) {
      long n = (long)K * N;
      cvt_t_bf16_kernel<<<dim3((unsigned)(n / 256)), dim3(256), 0, stream>>>(
          src + (long)i * n, dst + (long)i * n, K, N);
    }
  };
  cvt_t(wq, wqb, D_EMB, D_EMB);
  cvt_t(wk, wkb, D_EMB, D_EMB);
  cvt_t(wv, wvb, D_EMB, D_EMB);
  cvt_t(proj_w, wpb, D_EMB, D_EMB);
  cvt_t(mlp1_w, m1b, D_EMB, DFF);
  cvt_t(mlp2_w, m2b, DFF, D_EMB);

  // patch embed + cls + pos
  patch_embed_kernel<<<dim3(BATCH * NTOK), dim3(256), 0, stream>>>(
      x, conv_w, conv_b, cls_tok, pos_emb, t);

  auto gemm = [&](const unsigned short* A, const unsigned short* Wt, const float* bias,
                  const float* res, float* o32, unsigned short* o16, int Nc, int K,
                  float scale, int act) {
    dim3 g(MTOK / GMT, (unsigned)(Nc / GNT));
    gemm_bf16_kernel<<<g, dim3(256), 0, stream>>>(A, Wt, bias, res, o32, o16, Nc, K, scale, act);
  };

  const float qscale = 0.08838834764831845f;   // (D/H)^-0.5 = 128^-0.5
  for (int i = 0; i < NLAYER; i++) {
    layernorm_bf16_kernel<<<dim3(MTOK), dim3(256), 0, stream>>>(
        t, ln1_g + i * D_EMB, ln1_b + i * D_EMB, xn);
    long wo = (long)i * D_EMB * D_EMB;
    gemm(xn, wqb + wo, nullptr, nullptr, nullptr, qb, D_EMB, D_EMB, qscale, 0);
    gemm(xn, wkb + wo, nullptr, nullptr, nullptr, kb, D_EMB, D_EMB, 1.0f, 0);
    gemm(xn, wvb + wo, nullptr, nullptr, nullptr, vb, D_EMB, D_EMB, 1.0f, 0);
    attn_fused_kernel<<<dim3(17, NHEAD, BATCH), dim3(32), 0, stream>>>(qb, kb, vb, yb);
    gemm(yb, wpb + wo, proj_b + i * D_EMB, t, t, nullptr, D_EMB, D_EMB, 1.0f, 0);
    layernorm_bf16_kernel<<<dim3(MTOK), dim3(256), 0, stream>>>(
        t, ln2_g + i * D_EMB, ln2_b + i * D_EMB, xn);
    gemm(xn, m1b + (long)i * D_EMB * DFF, mlp1_b + i * DFF, nullptr, nullptr, hb,
         DFF, D_EMB, 1.0f, 1);
    gemm(hb, m2b + (long)i * DFF * D_EMB, mlp2_b + i * D_EMB, t, t, nullptr,
         D_EMB, DFF, 1.0f, 0);
  }

  extract_cls_kernel<<<dim3(BATCH), dim3(256), 0, stream>>>(t, out);
}
